// BMManager_74328704025133
// MI455X (gfx1250) — compile-verified, matrix operand order checked by reading, not run on hardware
//
#include <hip/hip_runtime.h>
#include <hip/hip_bf16.h>

// ---------------------------------------------------------------------------
// Problem constants (match reference)
// ---------------------------------------------------------------------------
#define BDIM 32
#define SDIM 4096
#define KDIM 1024
#define NDIM 512

// GEMM tiling: 128-thread blocks (4 waves). Block tile 64(M) x 128(N),
// K-blocked by 32. Each wave owns 2 M-tiles x 4 N-tiles = 8 accumulators.
#define BLOCK 128
#define TM 64
#define TN 128
#define TK 32
#define LDA 40     // LDS row stride in bf16 elems (80B: 16B-aligned, bank-friendly)
#define LDB 40
#define NKB (KDIM / TK)

typedef __attribute__((ext_vector_type(16))) __bf16 v16bf;
typedef __attribute__((ext_vector_type(8)))  float  v8f;

union Frag16 { v16bf v; uint4 q[2]; };   // 32 bytes: one WMMA 16-bit A/B operand
union Pack4  { __bf16 e[4]; uint2 u; };  // 8 bytes: 4 bf16

// ---------------------------------------------------------------------------
// Kernel 1: forward-fill segment index.
// idx[b,t] = max{ t' <= t : t'==0 || mask[b,t'-1] }  (cummax of sel)
// ---------------------------------------------------------------------------
__global__ __launch_bounds__(256)
void ffill_idx_kernel(const unsigned char* __restrict__ mask,
                      int* __restrict__ idx) {
  const int b   = blockIdx.x;
  const int tid = threadIdx.x;
  __shared__ int part[256];

  const int CH   = SDIM / 256;      // 16 contiguous t per thread
  const int base = tid * CH;

  int run = -1;
#pragma unroll
  for (int i = 0; i < CH; ++i) {
    const int t = base + i;
    const bool cond = (t == 0) || (mask[b * SDIM + t - 1] != 0);
    if (cond) run = t;
  }
  part[tid] = run;
  __syncthreads();

  for (int off = 1; off < 256; off <<= 1) {
    int v = part[tid];
    if (tid >= off) v = max(v, part[tid - off]);
    __syncthreads();
    part[tid] = v;
    __syncthreads();
  }
  int carry = (tid > 0) ? part[tid - 1] : -1;

  run = carry;
#pragma unroll
  for (int i = 0; i < CH; ++i) {
    const int t = base + i;
    const bool cond = (t == 0) || (mask[b * SDIM + t - 1] != 0);
    if (cond) run = t;
    idx[b * SDIM + t] = run;
  }
}

// ---------------------------------------------------------------------------
// Kernel 2: split fp32 W into bf16 hi + bf16 lo (hi = rne(w), lo = rne(w-hi)).
// ---------------------------------------------------------------------------
__global__ __launch_bounds__(256)
void wsplit_kernel(const float* __restrict__ W,
                   __bf16* __restrict__ Whi,
                   __bf16* __restrict__ Wlo) {
  const int i = (blockIdx.x * 256 + threadIdx.x) * 4;
  const float4 w = *reinterpret_cast<const float4*>(W + i);
  Pack4 hi, lo;
  const float f[4] = {w.x, w.y, w.z, w.w};
#pragma unroll
  for (int j = 0; j < 4; ++j) {
    const __bf16 h = (__bf16)f[j];
    hi.e[j] = h;
    lo.e[j] = (__bf16)(f[j] - (float)h);
  }
  *reinterpret_cast<uint2*>(Whi + i) = hi.u;
  *reinterpret_cast<uint2*>(Wlo + i) = lo.u;
}

// ---------------------------------------------------------------------------
// Kernel 3: fused gather + GEMM + bias, double-buffered LDS.
// out[b, t, :] = x[b, idx[b,t], :] @ W^T + bias
//
// Register-pressure discipline (avoid the round-2 scratch spills):
//   - A (HBM-streamed x): global loads issued at iteration start, held in
//     16 VGPRs across the WMMA block, converted+stored after compute.
//   - B (L2-resident W_hi/W_lo, 2MB total): loaded+stored to LDS only in the
//     post-compute phase; L2 latency is covered by co-resident waves.
// ---------------------------------------------------------------------------
__global__ __launch_bounds__(BLOCK)
void gather_gemm_kernel(const float* __restrict__ x,
                        const int* __restrict__ idx,
                        const __bf16* __restrict__ Whi,
                        const __bf16* __restrict__ Wlo,
                        const float* __restrict__ bias,
                        float* __restrict__ out) {
  const int mblk = blockIdx.x;   // 0..63   (S/TM)
  const int nblk = blockIdx.y;   // 0..3    (N/TN)
  const int bb   = blockIdx.z;   // 0..31   (B)
  const int tid  = threadIdx.x;
  const int lane = tid & 31;
  const int w    = tid >> 5;     // 0..3

  const int t0 = mblk * TM;
  const int n0 = nblk * TN;

  // double-buffered tiles: 2*(64+128)*40*2B*2(hi/lo) = 61440 B
  __shared__ __attribute__((aligned(16))) __bf16 As_hi[2][TM * LDA];
  __shared__ __attribute__((aligned(16))) __bf16 As_lo[2][TM * LDA];
  __shared__ __attribute__((aligned(16))) __bf16 Bs_hi[2][TN * LDB];
  __shared__ __attribute__((aligned(16))) __bf16 Bs_lo[2][TN * LDB];
  __shared__ int rowSrc[TM];

  if (tid < TM) rowSrc[tid] = idx[bb * SDIM + t0 + tid];
  __syncthreads();

  // ---- per-thread staging geometry (hoisted out of the K loop) ----
  // A: thread -> row tid/2, K-half (tid&1)*16 : 16 contiguous f32 (4 float4)
  const int arow = tid >> 1;
  const int akk  = (tid & 1) * 16;
  const float* aSrc =
      x + ((size_t)bb * SDIM + rowSrc[arow]) * KDIM + akk;   // one LDS read, then regs
  // B: thread -> full row 'tid' of the N-tile: 32 bf16 (4 uint4) for hi and lo
  const __bf16* bHiSrc = Whi + (size_t)(n0 + tid) * KDIM;
  const __bf16* bLoSrc = Wlo + (size_t)(n0 + tid) * KDIM;

  float4 ra[4];   // only A is register-staged across the compute phase

  auto g2rA = [&](int kb) {
#pragma unroll
    for (int s = 0; s < 4; ++s)
      ra[s] = *reinterpret_cast<const float4*>(aSrc + kb + s * 4);
  };
  auto storeA = [&](int buf) {
#pragma unroll
    for (int s = 0; s < 4; ++s) {
      Pack4 ph, pl;
      const float f[4] = {ra[s].x, ra[s].y, ra[s].z, ra[s].w};
#pragma unroll
      for (int j = 0; j < 4; ++j) {
        const __bf16 h = (__bf16)f[j];
        ph.e[j] = h;
        pl.e[j] = (__bf16)(f[j] - (float)h);
      }
      *reinterpret_cast<uint2*>(&As_hi[buf][arow * LDA + akk + s * 4]) = ph.u;
      *reinterpret_cast<uint2*>(&As_lo[buf][arow * LDA + akk + s * 4]) = pl.u;
    }
  };
  auto copyB = [&](int buf, int kb) {    // load->store, no cross-phase staging
#pragma unroll
    for (int s = 0; s < 4; ++s) {
      const uint4 h = *reinterpret_cast<const uint4*>(bHiSrc + kb + s * 8);
      const uint4 l = *reinterpret_cast<const uint4*>(bLoSrc + kb + s * 8);
      *reinterpret_cast<uint4*>(&Bs_hi[buf][tid * LDB + s * 8]) = h;
      *reinterpret_cast<uint4*>(&Bs_lo[buf][tid * LDB + s * 8]) = l;
    }
  };

  // ---- wave tiling: 2 M-tiles x 4 N-tiles per wave ----
  const int m0 = (w & 1) * 32;        // wave rows [m0, m0+32)
  const int nb = (w >> 1) * 64;       // wave cols [nb, nb+64)
  const int hl = lane >> 4;           // lane half: K sub-range (ISA A/B layout)
  const int ml = lane & 15;           // row (A) / col (B) within 16x16 tile

  v8f acc[2][4] = {};

  // prologue
  g2rA(0);
  storeA(0);
  copyB(0, 0);
  __syncthreads();

  for (int it = 0; it < NKB; ++it) {
    const int cur = it & 1;
    const int kb1 = (it + 1) * TK;
    const bool more = (kb1 < KDIM);
    if (more) g2rA(kb1);              // HBM loads overlap the WMMA block below

    // A fragments (ISA 7.12.2: lane=M, VGPR j holds K=(j>>2)*16 + h*8 + (j&3)*2)
    Frag16 ah[2], al[2];
#pragma unroll
    for (int mi = 0; mi < 2; ++mi) {
      const int am = m0 + mi * 16 + ml;
      ah[mi].q[0] = *reinterpret_cast<const uint4*>(&As_hi[cur][am * LDA + hl * 8]);
      ah[mi].q[1] = *reinterpret_cast<const uint4*>(&As_hi[cur][am * LDA + 16 + hl * 8]);
      al[mi].q[0] = *reinterpret_cast<const uint4*>(&As_lo[cur][am * LDA + hl * 8]);
      al[mi].q[1] = *reinterpret_cast<const uint4*>(&As_lo[cur][am * LDA + 16 + hl * 8]);
    }

#pragma unroll
    for (int j = 0; j < 4; ++j) {
      const int bn = nb + j * 16 + ml;         // B fragment: lane = N column
      Frag16 bh, bl;
      bh.q[0] = *reinterpret_cast<const uint4*>(&Bs_hi[cur][bn * LDB + hl * 8]);
      bh.q[1] = *reinterpret_cast<const uint4*>(&Bs_hi[cur][bn * LDB + 16 + hl * 8]);
      bl.q[0] = *reinterpret_cast<const uint4*>(&Bs_lo[cur][bn * LDB + hl * 8]);
      bl.q[1] = *reinterpret_cast<const uint4*>(&Bs_lo[cur][bn * LDB + 16 + hl * 8]);
#pragma unroll
      for (int mi = 0; mi < 2; ++mi) {
        acc[mi][j] = __builtin_amdgcn_wmma_f32_16x16x32_bf16(
            false, ah[mi].v, false, bh.v, (short)0, acc[mi][j], false, false);
        acc[mi][j] = __builtin_amdgcn_wmma_f32_16x16x32_bf16(
            false, ah[mi].v, false, bl.v, (short)0, acc[mi][j], false, false);
        acc[mi][j] = __builtin_amdgcn_wmma_f32_16x16x32_bf16(
            false, al[mi].v, false, bh.v, (short)0, acc[mi][j], false, false);
      }
    }

    if (more) {                // buf(1-cur) last read at it-1; barrier(it-1) protects
      storeA(1 - cur);
      copyB(1 - cur, kb1);     // L2-resident: short latency, hidden by other waves
    }
    __syncthreads();           // stores visible before it+1 compute
  }

  // ---- epilogue: C/D layout (VGPR r: lanes0-15 M=r, lanes16-31 M=r+8) ----
#pragma unroll
  for (int j = 0; j < 4; ++j) {
    const int n  = n0 + nb + j * 16 + ml;
    const float bv = bias[n];
#pragma unroll
    for (int mi = 0; mi < 2; ++mi) {
#pragma unroll
      for (int r = 0; r < 8; ++r) {
        const int m = t0 + m0 + mi * 16 + hl * 8 + r;
        out[((size_t)bb * SDIM + m) * NDIM + n] = acc[mi][j][r] + bv;
      }
    }
  }
}

// ---------------------------------------------------------------------------
// Launch
// ---------------------------------------------------------------------------
extern "C" void kernel_launch(void* const* d_in, const int* in_sizes, int n_in,
                              void* d_out, int out_size, void* d_ws, size_t ws_size,
                              hipStream_t stream) {
  const float*         x    = (const float*)d_in[0];
  const unsigned char* mask = (const unsigned char*)d_in[1]; // jnp.bool_: 1B/elem
  const float*         W    = (const float*)d_in[2];
  const float*         bias = (const float*)d_in[3];
  float*               out  = (float*)d_out;

  // workspace layout: idx (512KB) | W_hi (1MB) | W_lo (1MB)
  char* ws = (char*)d_ws;
  int*    idx = (int*)ws;
  __bf16* Whi = (__bf16*)(ws + (size_t)BDIM * SDIM * sizeof(int));
  __bf16* Wlo = Whi + (size_t)NDIM * KDIM;

  ffill_idx_kernel<<<BDIM, 256, 0, stream>>>(mask, idx);
  wsplit_kernel<<<(NDIM * KDIM) / (256 * 4), 256, 0, stream>>>(W, Whi, Wlo);

  dim3 grid(SDIM / TM, NDIM / TN, BDIM);
  gather_gemm_kernel<<<grid, BLOCK, 0, stream>>>(x, idx, Whi, Wlo, bias, out);
}
